// BitNetAttention_89489938580130
// MI455X (gfx1250) — matrix operand order repaired and checked by reading, not compiled, not run on hardware
//
#include <hip/hip_runtime.h>
#include <stdint.h>

// ---------------- problem constants ----------------
constexpr int kS  = 2048;           // sequence
constexpr int kB  = 4;              // batch
constexpr int kE  = 1024;           // embed
constexpr int kH  = 16;             // heads
constexpr int kHD = 64;             // head dim
constexpr int kN  = kB * kH;        // 64 batched heads
constexpr int kR  = kS * kB;        // 8192 rows for projections
constexpr float kEPS   = 1e-8f;
constexpr float kSCALE = 0.125f;    // HD^-0.5

// ---------------- workspace layout (bytes) ----------------
constexpr size_t WBYTES   = (size_t)kE * kE * 2;         // one bf16 weight: 2 MB
constexpr size_t OFF_SUMS = 0;                           // 6 floats (|Wq|,|Wk|,|Wv|,|Wo|,|x|,|attn|)
constexpr size_t OFF_QW   = 256;
constexpr size_t OFF_XBF  = OFF_QW + 4 * WBYTES;         // x/a   bf16 [kR,kE]
constexpr size_t XBYTES   = (size_t)kR * kE * 2;
constexpr size_t OFF_QH   = OFF_XBF + XBYTES;            // q bf16 [kN][kS][kHD]
constexpr size_t HBYTES   = (size_t)kN * kS * kHD * 2;
constexpr size_t OFF_KH   = OFF_QH + HBYTES;             // k bf16 [kN][kS][kHD]
constexpr size_t OFF_VT   = OFF_KH + HBYTES;             // vT bf16 [kN][kHD][kS]
constexpr size_t OFF_RMAX = OFF_VT + HBYTES;             // f32 [kN][kS]
constexpr size_t RBYTES   = (size_t)kN * kS * 4;
constexpr size_t OFF_RSUM = OFF_RMAX + RBYTES;
constexpr size_t OFF_ATTN = OFF_RSUM + RBYTES;           // f32 attn [kR,kE]
constexpr size_t ABYTES   = (size_t)kR * kE * 4;
constexpr size_t OFF_ABF  = OFF_ATTN + ABYTES;           // bf16 attn/a2 [kR,kE]

// ---------------- WMMA plumbing ----------------
typedef __bf16 bf16_t;
typedef bf16_t v16bf __attribute__((ext_vector_type(16)));
typedef float  v8f   __attribute__((ext_vector_type(8)));
typedef unsigned int v4u __attribute__((ext_vector_type(4)));
typedef int v8i __attribute__((ext_vector_type(8)));
typedef int v4i __attribute__((ext_vector_type(4)));

union ABOp { uint4 q[2]; v16bf v; };

// A operand (16x32, bf16): lane holds row (lane&15); lanes<16: K 0..7 & 16..23,
// lanes>=16: K 8..15 & 24..31  -> two 16B loads at kbase and kbase+16.
__device__ __forceinline__ v16bf load_a16(const unsigned short* base, int ld,
                                          int row, int k0, int lane) {
  ABOp op;
  const unsigned short* p = base + (size_t)row * ld + k0 + ((lane >> 4) << 3);
  op.q[0] = *reinterpret_cast<const uint4*>(p);
  op.q[1] = *reinterpret_cast<const uint4*>(p + 16);
  return op.v;
}

// B operand (32x16, bf16) loaded from B^T storage: lane holds column (lane&15),
// lanes<16: K 0..7 & 16..23, lanes>=16: K 8..15 & 24..31 -> two 16B loads.
__device__ __forceinline__ v16bf load_b16(const unsigned short* baseT, int ld,
                                          int col, int k0, int lane) {
  ABOp op;
  const unsigned short* p = baseT + (size_t)col * ld + k0 + ((lane >> 4) << 3);
  op.q[0] = *reinterpret_cast<const uint4*>(p);
  op.q[1] = *reinterpret_cast<const uint4*>(p + 16);
  return op.v;
}

__device__ __forceinline__ v8f wmma_bf16(v16bf a, v16bf b, v8f c) {
  return __builtin_amdgcn_wmma_f32_16x16x32_bf16(false, a, false, b,
                                                 (short)0, c, false, false);
}

__device__ __forceinline__ unsigned short f2bf(float f) {
  unsigned int u = __float_as_uint(f);
  u += 0x7FFFu + ((u >> 16) & 1u);              // round-to-nearest-even
  return (unsigned short)(u >> 16);
}

// ---------------- Tensor Data Mover: stage a 64x32 bf16 weight chunk into LDS ----
// D# 2D tile: tile_dim0=32 elems along a row (contiguous), tile_dim1=64 rows,
// row stride = kE elements, data_size=2B, landing contiguously (4KB) at lds_off.
__device__ __forceinline__ void tdm_load_b_tile(const unsigned short* gptr,
                                                unsigned lds_off) {
  unsigned long long ga = (unsigned long long)(uintptr_t)gptr;
  v4u g0;
  g0[0] = 1u;                                             // count=1, user mode
  g0[1] = lds_off;                                        // LDS byte address
  g0[2] = (unsigned)(ga & 0xFFFFFFFFu);                   // global_addr[31:0]
  g0[3] = (unsigned)((ga >> 32) & 0x1FFFFFFu) | (2u << 30); // addr[56:32] | type=2
  v8i g1;
  g1[0] = (int)(1u << 16);                                // data_size=1 -> 2 bytes
  g1[1] = (int)(((unsigned)kE & 0xFFFFu) << 16);          // tensor_dim0[15:0]
  g1[2] = (int)((((unsigned)kE >> 16) & 0xFFFFu) |        // tensor_dim0[31:16]
                (((unsigned)kE & 0xFFFFu) << 16));        // tensor_dim1[15:0]
  g1[3] = (int)((((unsigned)kE >> 16) & 0xFFFFu) |        // tensor_dim1[31:16]
                (32u << 16));                             // tile_dim0 = 32
  g1[4] = (int)(64u);                                     // tile_dim1=64, tile_dim2=0
  g1[5] = (int)kE;                                        // tensor_dim0_stride[31:0]
  g1[6] = 0;                                              // stride0 hi | stride1 lo
  g1[7] = 0;
  v4i z4 = {0, 0, 0, 0};
#if __clang_major__ >= 23
  v8i z8 = {0, 0, 0, 0, 0, 0, 0, 0};
  __builtin_amdgcn_tensor_load_to_lds(g0, g1, z4, z4, z8, 0);
#else
  __builtin_amdgcn_tensor_load_to_lds(g0, g1, z4, z4, 0);
#endif
}

// ---------------- kernel 1: abs-sum reduction ----------------
__global__ __launch_bounds__(256) void bn_absum(const float* __restrict__ x,
                                                size_t n, float* __restrict__ sums,
                                                int slot) {
  __shared__ float red[256];
  float s = 0.f;
  for (size_t i = (size_t)blockIdx.x * blockDim.x + threadIdx.x; i < n;
       i += (size_t)gridDim.x * blockDim.x)
    s += fabsf(x[i]);
  red[threadIdx.x] = s;
  __syncthreads();
  for (int o = 128; o > 0; o >>= 1) {
    if ((int)threadIdx.x < o) red[threadIdx.x] += red[threadIdx.x + o];
    __syncthreads();
  }
  if (threadIdx.x == 0) atomicAdd(&sums[slot], red[0]);
}

// ---------------- kernel 2: ternary weight quantization (store +/-1,0 as bf16) ----
__global__ __launch_bounds__(256) void bn_quant_w(const float* __restrict__ W,
                                                  unsigned short* __restrict__ qw,
                                                  const float* __restrict__ sums,
                                                  int slot, float invn) {
  const float ws  = fmaxf(sums[slot] * invn, kEPS);
  const float inv = 1.f / ws;
  for (size_t i = (size_t)blockIdx.x * blockDim.x + threadIdx.x;
       i < (size_t)kE * kE; i += (size_t)gridDim.x * blockDim.x) {
    float wn = W[i] * inv;
    float q  = (fabsf(wn) > 0.5f) ? (wn > 0.f ? 1.f : -1.f) : 0.f;
    qw[i] = f2bf(q);
  }
}

// ---------------- kernel 3: activation quantization x -> bf16(x / a_scale) -------
__global__ __launch_bounds__(256) void bn_quant_x(const float* __restrict__ x,
                                                  unsigned short* __restrict__ xb,
                                                  const float* __restrict__ sums,
                                                  int slot, float invn, size_t n) {
  const float a   = fmaxf(sums[slot] * invn, kEPS);
  const float inv = 1.f / a;
  for (size_t i = (size_t)blockIdx.x * blockDim.x + threadIdx.x; i < n;
       i += (size_t)gridDim.x * blockDim.x)
    xb[i] = f2bf(x[i] * inv);
}

// ---------------- kernel 4: BitNet GEMM via WMMA + TDM-staged weights ----------
// y = a_scale * (w_scale * (xn @ qw^T) + bias). 8 waves/WG, 16x64 strip per wave.
// Weight chunk (64 cols x 32 K, 4KB) staged in LDS by the Tensor Data Mover,
// double-buffered so the DMA for step k+1 overlaps the WMMAs of step k.
// mode 0: store bf16 q/k head layout [n][s][hd]; mode 2: vT [n][hd][s]; mode 3: f32 out.
__global__ __launch_bounds__(256) void bn_gemm(const unsigned short* __restrict__ xbf,
                                               const unsigned short* __restrict__ qw,
                                               const float* __restrict__ bias,
                                               const float* __restrict__ sums,
                                               int wslot, int aslot,
                                               float inv_w, float inv_a,
                                               unsigned short* __restrict__ outb,
                                               float* __restrict__ outf, int mode) {
  __shared__ __align__(16) unsigned short bstage[2][64 * 32];
  const int lane = threadIdx.x & 31;
  const int wave = threadIdx.x >> 5;
  const int r0   = blockIdx.x * 128 + wave * 16;
  const int n0   = blockIdx.y * 64;
  const float wsc = fmaxf(sums[wslot] * inv_w, kEPS);
  const float asc = fmaxf(sums[aslot] * inv_a, kEPS);

  const unsigned lds0 = (unsigned)(uintptr_t)&bstage[0][0];
  constexpr unsigned kSlab = 64 * 32 * 2;                 // 4KB per buffer
  const unsigned short* gB = qw + (size_t)n0 * kE;        // rows n0..n0+63

  v8f acc[4];
#pragma unroll
  for (int t = 0; t < 4; t++) acc[t] = v8f{};

  // prologue: DMA first weight chunk, prefetch first A operand
  if (threadIdx.x == 0) tdm_load_b_tile(gB, lds0);
  const int arow = r0 + (lane & 15);
  v16bf a = load_a16(xbf, kE, arow, 0, lane);
  __builtin_amdgcn_s_wait_tensorcnt((short)0);
  __syncthreads();

  for (int k0 = 0; k0 < kE; k0 += 32) {
    const int cur = (k0 >> 5) & 1;
    // kick off DMA for next chunk into the other buffer (overlaps WMMAs below)
    if (k0 + 32 < kE && threadIdx.x == 0)
      tdm_load_b_tile(gB + (k0 + 32), lds0 + (unsigned)(cur ^ 1) * kSlab);
    // prefetch next A operand into registers (overlaps WMMAs below)
    v16bf a_next = a;
    if (k0 + 32 < kE) a_next = load_a16(xbf, kE, arow, k0 + 32, lane);

    const unsigned short* bl = &bstage[cur][0];
#pragma unroll
    for (int t = 0; t < 4; t++) {
      v16bf b = load_b16(bl, 32, t * 16 + (lane & 15), 0, lane);
      acc[t] = wmma_bf16(a, b, acc[t]);
    }
    a = a_next;
    __builtin_amdgcn_s_wait_tensorcnt((short)0);
    __syncthreads();                                     // drains DS reads too
  }

  const int rbase = (lane >> 4) << 3;
#pragma unroll
  for (int t = 0; t < 4; t++) {
    const int o   = n0 + t * 16 + (lane & 15);
    const float bc = bias[o];
#pragma unroll
    for (int i = 0; i < 8; i++) {
      const int r = r0 + i + rbase;
      const float y = asc * (wsc * acc[t][i] + bc);
      if (mode == 3) {
        outf[(size_t)r * kE + o] = y;
      } else {
        const int s = r >> 2;         // r = s*kB + b, kB == 4
        const int b = r & 3;
        const int h = o >> 6, d = o & 63;
        const int n = b * kH + h;
        const size_t idx = (mode == 2) ? ((size_t)(n * kHD + d) * kS + s)
                                       : ((size_t)(n * kS + s) * kHD + d);
        outb[idx] = f2bf(y);
      }
    }
  }
}

// ---------------- kernel 5: attention core (one wave per head x 16-row q tile) ----
__global__ __launch_bounds__(32) void bn_attn(const unsigned short* __restrict__ qb,
                                              const unsigned short* __restrict__ kb,
                                              const unsigned short* __restrict__ vtb,
                                              float* __restrict__ rowmax,
                                              float* __restrict__ rowsum,
                                              float* __restrict__ attn,
                                              float* __restrict__ sums) {
  __shared__ __align__(16) unsigned short pL[16 * 32];   // staged P tile (bf16)
  const int lane = threadIdx.x;
  const int n    = blockIdx.x >> 7;                      // 128 q tiles per head
  const int sq0  = (blockIdx.x & 127) * 16;
  const unsigned short* qh = qb  + (size_t)n * kS * kHD;
  const unsigned short* kh = kb  + (size_t)n * kS * kHD;
  const unsigned short* vh = vtb + (size_t)n * kHD * kS;
  const int rbase = (lane >> 4) << 3;
  const int aq    = sq0 + (lane & 15);

  const v16bf a0 = load_a16(qh, kHD, aq, 0, lane);       // q A-operand, K 0..31
  const v16bf a1 = load_a16(qh, kHD, aq, 32, lane);      // q A-operand, K 32..63

  // ---- sweep 1: row max ----
  float m[8];
#pragma unroll
  for (int i = 0; i < 8; i++) m[i] = -3.0e38f;
  for (int sk0 = 0; sk0 < kS; sk0 += 16) {
    if (sk0 + 16 < kS)
      __builtin_prefetch(kh + (size_t)(sk0 + 16 + (lane & 15)) * kHD, 0, 0);
    v8f c = v8f{};
    c = wmma_bf16(a0, load_b16(kh, kHD, sk0 + (lane & 15), 0, lane), c);
    c = wmma_bf16(a1, load_b16(kh, kHD, sk0 + (lane & 15), 32, lane), c);
#pragma unroll
    for (int i = 0; i < 8; i++) m[i] = fmaxf(m[i], c[i] * kSCALE);
  }
#pragma unroll
  for (int i = 0; i < 8; i++)
    for (int o = 8; o >= 1; o >>= 1) m[i] = fmaxf(m[i], __shfl_xor(m[i], o, 16));
  if ((lane & 15) == 0)
    for (int i = 0; i < 8; i++)
      rowmax[(size_t)n * kS + sq0 + rbase + i] = m[i];

  // ---- sweep 2: exp / rowsum / P @ V ----
  v8f oacc[4];
#pragma unroll
  for (int t = 0; t < 4; t++) oacc[t] = v8f{};
  float rs[8];
#pragma unroll
  for (int i = 0; i < 8; i++) rs[i] = 0.f;

  for (int sk0 = 0; sk0 < kS; sk0 += 32) {
#pragma unroll
    for (int half = 0; half < 2; half++) {
      const int skt = sk0 + half * 16;
      v8f c = v8f{};
      c = wmma_bf16(a0, load_b16(kh, kHD, skt + (lane & 15), 0, lane), c);
      c = wmma_bf16(a1, load_b16(kh, kHD, skt + (lane & 15), 32, lane), c);
#pragma unroll
      for (int i = 0; i < 8; i++) {
        const float p = __expf(c[i] * kSCALE - m[i]);
        rs[i] += p;
        pL[(i + rbase) * 32 + half * 16 + (lane & 15)] = f2bf(p);
      }
    }
    asm volatile("s_wait_dscnt 0x0" ::: "memory");       // P stores visible
    const v16bf pa = load_a16(pL, 32, lane & 15, 0, lane);
#pragma unroll
    for (int t = 0; t < 4; t++) {
      v16bf bv = load_b16(vh, kS, t * 16 + (lane & 15), sk0, lane);
      oacc[t] = wmma_bf16(pa, bv, oacc[t]);
    }
    asm volatile("s_wait_dscnt 0x0" ::: "memory");       // P reads done before overwrite
  }

#pragma unroll
  for (int i = 0; i < 8; i++)
    for (int o = 8; o >= 1; o >>= 1) rs[i] += __shfl_xor(rs[i], o, 16);
  if ((lane & 15) == 0)
    for (int i = 0; i < 8; i++)
      rowsum[(size_t)n * kS + sq0 + rbase + i] = rs[i];

  // ---- normalize + store attn (f32) + |attn| partial sum ----
  const int b = n / kH, h = n % kH;
  float sumabs = 0.f;
#pragma unroll
  for (int t = 0; t < 4; t++) {
#pragma unroll
    for (int i = 0; i < 8; i++) {
      const int s = sq0 + i + rbase;
      const int d = t * 16 + (lane & 15);
      const float val = oacc[t][i] / rs[i];
      attn[((size_t)s * kB + b) * kE + h * kHD + d] = val;
      sumabs += fabsf(val);
    }
  }
  for (int o = 16; o >= 1; o >>= 1) sumabs += __shfl_xor(sumabs, o, 32);
  if (lane == 0) atomicAdd(&sums[5], sumabs);
}

// ---------------- kernel 6: head-averaged attention weights ----------------
__global__ __launch_bounds__(256) void bn_attnw(const unsigned short* __restrict__ qb,
                                                const unsigned short* __restrict__ kb,
                                                const float* __restrict__ rowmax,
                                                const float* __restrict__ rowsum,
                                                float* __restrict__ aw) {
  __shared__ float rmaxL[16 * 16];
  __shared__ float rinvL[16 * 16];
  const int b   = blockIdx.y;
  const int sq0 = blockIdx.x * 16;
  const int tid = threadIdx.x;
  {
    const int h = tid >> 4, row = tid & 15;
    const int n = b * kH + h;
    rmaxL[tid] = rowmax[(size_t)n * kS + sq0 + row];
    rinvL[tid] = (1.0f / (float)kH) / rowsum[(size_t)n * kS + sq0 + row];
  }
  __syncthreads();

  const int lane = tid & 31, wave = tid >> 5;
  const int rbase = (lane >> 4) << 3;
  for (int kt = wave; kt < kS / 16; kt += 8) {
    const int sk0 = kt * 16;
    v8f acc = v8f{};
    for (int h = 0; h < kH; h++) {
      const int n = b * kH + h;
      const unsigned short* qh = qb + (size_t)n * kS * kHD;
      const unsigned short* kh = kb + (size_t)n * kS * kHD;
      v8f c = v8f{};
      c = wmma_bf16(load_a16(qh, kHD, sq0 + (lane & 15), 0, lane),
                    load_b16(kh, kHD, sk0 + (lane & 15), 0, lane), c);
      c = wmma_bf16(load_a16(qh, kHD, sq0 + (lane & 15), 32, lane),
                    load_b16(kh, kHD, sk0 + (lane & 15), 32, lane), c);
#pragma unroll
      for (int i = 0; i < 8; i++) {
        const int row = i + rbase;
        acc[i] += __expf(c[i] * kSCALE - rmaxL[h * 16 + row]) * rinvL[h * 16 + row];
      }
    }
#pragma unroll
    for (int i = 0; i < 8; i++) {
      const int row = i + rbase;
      aw[((size_t)b * kS + sq0 + row) * kS + sk0 + (lane & 15)] = acc[i];
    }
  }
}

// ---------------- host-side launch ----------------
extern "C" void kernel_launch(void* const* d_in, const int* in_sizes, int n_in,
                              void* d_out, int out_size, void* d_ws, size_t ws_size,
                              hipStream_t stream) {
  (void)in_sizes; (void)n_in; (void)out_size; (void)ws_size;
  const float* query = (const float*)d_in[0];
  const float* W[4]  = {(const float*)d_in[1], (const float*)d_in[3],
                        (const float*)d_in[5], (const float*)d_in[7]};
  const float* bias[4] = {(const float*)d_in[2], (const float*)d_in[4],
                          (const float*)d_in[6], (const float*)d_in[8]};

  char* ws = (char*)d_ws;
  float*          sums   = (float*)(ws + OFF_SUMS);
  unsigned short* qw0    = (unsigned short*)(ws + OFF_QW);
  unsigned short* xbf    = (unsigned short*)(ws + OFF_XBF);
  unsigned short* q_bf   = (unsigned short*)(ws + OFF_QH);
  unsigned short* k_bf   = (unsigned short*)(ws + OFF_KH);
  unsigned short* vT_bf  = (unsigned short*)(ws + OFF_VT);
  float*          rowmax = (float*)(ws + OFF_RMAX);
  float*          rowsum = (float*)(ws + OFF_RSUM);
  float*          attn   = (float*)(ws + OFF_ATTN);
  unsigned short* a_bf   = (unsigned short*)(ws + OFF_ABF);

  float* out_f  = (float*)d_out;                       // [kS,kB,kE]
  float* out_aw = out_f + (size_t)kR * kE;             // [kB,kS,kS]

  const float invW = 1.0f / ((float)kE * (float)kE);
  const float invX = 1.0f / ((float)kR * (float)kE);

  hipMemsetAsync(ws + OFF_SUMS, 0, 256, stream);

  // abs-mean sums: 4 weights + input activations
  for (int i = 0; i < 4; i++)
    bn_absum<<<dim3(256), dim3(256), 0, stream>>>(W[i], (size_t)kE * kE, sums, i);
  bn_absum<<<dim3(512), dim3(256), 0, stream>>>(query, (size_t)kR * kE, sums, 4);

  // quantize weights (ternary, bf16) and activations
  for (int i = 0; i < 4; i++)
    bn_quant_w<<<dim3(512), dim3(256), 0, stream>>>(
        W[i], qw0 + (size_t)i * kE * kE, sums, i, invW);
  bn_quant_x<<<dim3(1024), dim3(256), 0, stream>>>(query, xbf, sums, 4, invX,
                                                   (size_t)kR * kE);

  // Q/K/V projections (WMMA bf16 GEMMs, TDM-staged weights)
  dim3 ggrid(kR / 128, kE / 64);
  bn_gemm<<<ggrid, dim3(256), 0, stream>>>(xbf, qw0 + 0 * (size_t)kE * kE, bias[0],
                                           sums, 0, 4, invW, invX, q_bf, nullptr, 0);
  bn_gemm<<<ggrid, dim3(256), 0, stream>>>(xbf, qw0 + 1 * (size_t)kE * kE, bias[1],
                                           sums, 1, 4, invW, invX, k_bf, nullptr, 0);
  bn_gemm<<<ggrid, dim3(256), 0, stream>>>(xbf, qw0 + 2 * (size_t)kE * kE, bias[2],
                                           sums, 2, 4, invW, invX, vT_bf, nullptr, 2);

  // attention core: 64 heads x 128 q-tiles, one wave each
  bn_attn<<<dim3(kN * (kS / 16)), dim3(32), 0, stream>>>(q_bf, k_bf, vT_bf, rowmax,
                                                         rowsum, attn, sums);

  // quantize attention output for the final BitNet GEMM
  bn_quant_x<<<dim3(1024), dim3(256), 0, stream>>>(attn, a_bf, sums, 5, invX,
                                                   (size_t)kR * kE);

  // head-averaged attention weights
  bn_attnw<<<dim3(kS / 16, kB), dim3(256), 0, stream>>>(q_bf, k_bf, rowmax, rowsum,
                                                        out_aw);

  // output projection -> f32 out
  bn_gemm<<<ggrid, dim3(256), 0, stream>>>(a_bf, qw0 + 3 * (size_t)kE * kE, bias[3],
                                           sums, 3, 5, invW, invX, nullptr, out_f, 3);
}
